// PairRepulsionSwitch_50036368999024
// MI455X (gfx1250) — matrix operand order repaired and checked by reading, not compiled, not run on hardware
//
#include <hip/hip_runtime.h>
#include <hip/hip_bf16.h>
#include <math.h>

typedef __attribute__((ext_vector_type(16))) _Float16 v16h;
typedef __attribute__((ext_vector_type(8)))  float    v8f;
typedef __attribute__((ext_vector_type(4)))  float    f32x4;
typedef __attribute__((ext_vector_type(4)))  int      i32x4;

#define KE_C   14.3996454784255f
#define A0_C   0.52917721092f
#define LOG2E  1.4426950408889634f

// ---------------------------------------------------------------------------
// Kernel 1: zero the output accumulator (harness poisons d_out with 0xAA).
// ---------------------------------------------------------------------------
__global__ void zero_kernel(float* __restrict__ out, int n) {
    int i = blockIdx.x * blockDim.x + threadIdx.x;
    if (i < n) out[i] = 0.0f;
}

// ---------------------------------------------------------------------------
// Kernel 2: Z[node] = node_attrs @ atomic_numbers via v_wmma_f32_16x16x32_f16.
// One wave32 handles 16 nodes: A = 16x32 f16 (one-hot rows, K zero-padded),
// B = 32x16 f16 with atomic_numbers replicated down every column, so every
// column of D = A*B holds the same Z vector. Also precomputes Z^0.23 so the
// 6.4M-edge hot loop never touches powf (64x fewer pow evals).
//
// A layout (16-bit, 16x32 MxK): lanes 0-15 -> row M=lane, halves 0-7 = K0-7,
// halves 8-15 = K16-23; lanes 16-31 -> row M=lane-16, halves 0-7 = K8-15,
// halves 8-15 = K24-31.
// B layout (16-bit, 32x16 KxN): lanes 0-15 hold K0-7, K16-23 for column N=lane,
// lanes 16-31 hold K8-15, K24-31 for column N=lane-16.
// D layout (f32 16x16): VGPR j, lanes 0-15 -> M=j, lanes 16-31 -> M=8+j.
// ---------------------------------------------------------------------------
__global__ __launch_bounds__(32) void znode_wmma_kernel(
    const float* __restrict__ attrs,     // [n_nodes, n_elem]
    const float* __restrict__ atn,       // [n_elem]
    float* __restrict__ Z,               // [n_nodes] out
    float* __restrict__ Zp,              // [n_nodes] out: Z^0.23
    int n_nodes, int n_elem)
{
    const int lane = (int)threadIdx.x;   // 0..31
    const int base = (int)blockIdx.x * 16;
    const int row  = base + (lane & 15);
    const bool hi  = (lane >= 16);

    v16h a, b;
#pragma unroll
    for (int i = 0; i < 16; ++i) {
        // K index this half-slot holds in the A operand
        int ka = hi ? ((i < 8) ? (i + 8) : (i + 16))
                    : ((i < 8) ?  i      : (i + 8));
        float av = (ka < n_elem && row < n_nodes) ? attrs[row * n_elem + ka] : 0.0f;
        a[i] = (_Float16)av;
        // K index this half-slot holds in the B operand (column-replicated)
        int kb = hi ? ((i < 8) ? (i + 8) : (i + 16))
                    : ((i < 8) ?  i      : (i + 8));
        float bv = (kb < n_elem) ? atn[kb] : 0.0f;
        b[i] = (_Float16)bv;
    }

    v8f c = {};
    c = __builtin_amdgcn_wmma_f32_16x16x32_f16(
        /*neg_a=*/false, a, /*neg_b=*/false, b,
        /*c_mod=*/(short)0, c, /*reuse_a=*/false, /*reuse_b=*/false);

    // All columns identical; lane 0 owns rows base+0..7, lane 16 rows base+8..15.
    if ((lane & 15) == 0) {
        const int m0 = hi ? 8 : 0;
#pragma unroll
        for (int j = 0; j < 8; ++j) {
            int r = base + m0 + j;
            if (r < n_nodes) {
                float z = c[j];
                Z[r]  = z;
                Zp[r] = __powf(z, 0.23f);   // z >= 1 always (atomic numbers)
            }
        }
    }
}

// ---------------------------------------------------------------------------
// Per-edge physics. Raw v_rcp_f32 / v_exp_f32 (both ~1 ulp) keep the VALU
// count well under the HBM roofline (~50 -> ~25 ops/edge).
// ---------------------------------------------------------------------------
__device__ __forceinline__ void edge_one(
    float len, int si, int di,
    const float* __restrict__ Z, const float* __restrict__ Zp,
    float inv_rmax, float inv_a0, float* __restrict__ out)
{
    float r    = fmaxf(len, 0.2f);                 // R_MIN
    float Zi   = Z[si];
    float Zj   = Z[di];
    float rinv = __builtin_amdgcn_rcpf(r);         // v_rcp_f32

    // ZBL: x = r / a, a = 0.88534*a0/(Zi^.23+Zj^.23); xm = -log2(e)*x shared
    float x   = r * (Zp[si] + Zp[di]) * inv_a0;
    float xm  = -LOG2E * x;
    float phi = 0.1818f  * __builtin_amdgcn_exp2f(3.2f    * xm)
              + 0.5099f  * __builtin_amdgcn_exp2f(0.9423f * xm)
              + 0.2802f  * __builtin_amdgcn_exp2f(0.4029f * xm)
              + 0.02817f * __builtin_amdgcn_exp2f(0.2016f * xm);

    // polynomial cutoff (1 - clamp(r/rmax,0,1))^6 ; r >= 0.2 so lower clamp moot
    float xc  = fminf(r * inv_rmax, 1.0f);
    float omx = 1.0f - xc;
    float o2  = omx * omx;
    float cut = o2 * o2 * o2;

    float Vz = KE_C * Zi * Zj * phi * rinv * cut;  // SCALE == 1

    // C12 * r^-12 with poly cutoff + smoothstep switch at rc=1.5, w=0.2
    float ri2 = rinv * rinv;
    float ri4 = ri2 * ri2;
    float V12 = 1e-4f * (ri4 * ri4 * ri4) * cut;
    float t   = fminf(fmaxf((1.5f - r) * 5.0f, 0.0f), 1.0f);
    V12 *= t * t * (3.0f - 2.0f * t);

    float q = 0.25f * (Vz + V12);
    atomicAdd(out + si, q);                        // global_atomic_add_f32, no return
    atomicAdd(out + di, q);
}

// ---------------------------------------------------------------------------
// Kernel 3a: vectorized streaming edge pass (n_edges % 4 == 0 path).
// b128 non-temporal loads for the 77 MB single-pass stream (keeps L2 free
// for the Z tables + output accumulator); 4 edges per thread.
// ---------------------------------------------------------------------------
__global__ __launch_bounds__(256) void edge_kernel_vec4(
    const f32x4* __restrict__ lengths4,
    const i32x4* __restrict__ src4,
    const i32x4* __restrict__ dst4,
    const float* __restrict__ Z,
    const float* __restrict__ Zp,
    const float* __restrict__ rmax_p,
    float* __restrict__ out,
    int n_vec)
{
    const float inv_rmax = 1.0f / rmax_p[0];
    const float inv_a0   = 1.0f / (0.88534f * A0_C);

    const int stride = (int)(gridDim.x * blockDim.x);
    for (int v = (int)(blockIdx.x * blockDim.x + threadIdx.x); v < n_vec; v += stride) {
        f32x4 L = __builtin_nontemporal_load(&lengths4[v]);   // global_load_b128 NT
        i32x4 S = __builtin_nontemporal_load(&src4[v]);
        i32x4 D = __builtin_nontemporal_load(&dst4[v]);
#pragma unroll
        for (int j = 0; j < 4; ++j)
            edge_one(L[j], S[j], D[j], Z, Zp, inv_rmax, inv_a0, out);
    }
}

// ---------------------------------------------------------------------------
// Kernel 3b: scalar edge pass (tail / unaligned fallback).
// ---------------------------------------------------------------------------
__global__ __launch_bounds__(256) void edge_kernel_scalar(
    const float* __restrict__ lengths,
    const int*   __restrict__ src_idx,
    const int*   __restrict__ dst_idx,
    const float* __restrict__ Z,
    const float* __restrict__ Zp,
    const float* __restrict__ rmax_p,
    float* __restrict__ out,
    int e0, int n_edges)
{
    const float inv_rmax = 1.0f / rmax_p[0];
    const float inv_a0   = 1.0f / (0.88534f * A0_C);

    const int stride = (int)(gridDim.x * blockDim.x);
    for (int e = e0 + (int)(blockIdx.x * blockDim.x + threadIdx.x); e < n_edges; e += stride)
        edge_one(lengths[e], src_idx[e], dst_idx[e], Z, Zp, inv_rmax, inv_a0, out);
}

// ---------------------------------------------------------------------------
// Launch
// ---------------------------------------------------------------------------
extern "C" void kernel_launch(void* const* d_in, const int* in_sizes, int n_in,
                              void* d_out, int out_size, void* d_ws, size_t ws_size,
                              hipStream_t stream) {
    const float* lengths = (const float*)d_in[0];   // [n_edges] f32
    const float* attrs   = (const float*)d_in[1];   // [n_nodes, n_elem] f32
    const int*   eidx    = (const int*)  d_in[2];   // [2, n_edges] i32
    const float* atn     = (const float*)d_in[3];   // [n_elem] f32
    const float* rmax    = (const float*)d_in[4];   // [1] f32

    const int n_edges = in_sizes[2] / 2;
    const int n_elem  = in_sizes[3];
    const int n_nodes = in_sizes[1] / n_elem;

    float* Z   = (float*)d_ws;           // [n_nodes]
    float* Zp  = Z + n_nodes;            // [n_nodes]
    float* out = (float*)d_out;

    // 1) zero output accumulator
    zero_kernel<<<(out_size + 255) / 256, 256, 0, stream>>>(out, out_size);

    // 2) WMMA precompute of Z and Z^0.23 per node (one wave per 16 nodes)
    znode_wmma_kernel<<<(n_nodes + 15) / 16, 32, 0, stream>>>(
        attrs, atn, Z, Zp, n_nodes, n_elem);

    // 3) streaming edge pass with scatter-add
    const int threads = 256;
    if ((n_edges & 3) == 0) {
        const int n_vec = n_edges >> 2;
        int blocks = (n_vec + threads - 1) / threads;
        if (blocks > 16384) blocks = 16384;
        edge_kernel_vec4<<<blocks, threads, 0, stream>>>(
            (const f32x4*)lengths, (const i32x4*)eidx,
            (const i32x4*)(eidx + n_edges), Z, Zp, rmax, out, n_vec);
    } else {
        int blocks = (n_edges + threads - 1) / threads;
        if (blocks > 24576) blocks = 24576;
        edge_kernel_scalar<<<blocks, threads, 0, stream>>>(
            lengths, eidx, eidx + n_edges, Z, Zp, rmax, out, 0, n_edges);
    }
}
